// PAM_Module_12429635355269
// MI455X (gfx1250) — matrix operand order repaired and checked by reading, not compiled, not run on hardware
//
#include <hip/hip_runtime.h>

#define BB 4
#define CC 512
#define DD 64
#define NN 4096   // 64*64
#define KB 64     // keys per flash-attention block

typedef __bf16 bf16;
typedef __bf16 v16bf __attribute__((ext_vector_type(16)));
typedef __bf16 v8bf  __attribute__((ext_vector_type(8)));
typedef float  v8f   __attribute__((ext_vector_type(8)));
typedef float  v4f   __attribute__((ext_vector_type(4)));

union V16 { v16bf v; v8bf h[2]; };

__device__ __forceinline__ float fast_exp2(float x) {
  return __builtin_amdgcn_exp2f(x);   // v_exp_f32 (native base-2)
}

// ---------------------------------------------------------------------------
// Kernel 1: x (B,C,N) f32 -> xT (B,N,C) bf16   (LDS tile transpose)
// ---------------------------------------------------------------------------
__global__ void xpose_bf16(const float* __restrict__ x, bf16* __restrict__ xT) {
  __shared__ float tile[16][17];
  int b  = blockIdx.z;
  int c0 = blockIdx.x * 16, n0 = blockIdx.y * 16;
  int tx = threadIdx.x, ty = threadIdx.y;
  const float* xb = x + (size_t)b * CC * NN;
  tile[ty][tx] = xb[(size_t)(c0 + ty) * NN + n0 + tx];
  __syncthreads();
  bf16* xtb = xT + (size_t)b * NN * CC;
  xtb[(size_t)(n0 + ty) * CC + c0 + tx] = (bf16)tile[tx][ty];
}

// ---------------------------------------------------------------------------
// Kernel 2: Out(M x N) = ascale * (Wmat(M x C) * X(C x N) + bias), bf16 out.
// One wave computes a 16(M) x 64(N) tile. ascale = log2(e) for the Q
// projection so the attention softmax can run in the base-2 domain.
// ---------------------------------------------------------------------------
__global__ __launch_bounds__(32) void proj_gemm(
    const float* __restrict__ Wmat, const float* __restrict__ bias,
    const bf16* __restrict__ xT, bf16* __restrict__ out,
    int M, long sm, long sn, float ascale) {
  int lane = threadIdx.x;
  int g = lane >> 4, li = lane & 15;
  int mt = blockIdx.x, nb = blockIdx.y, b = blockIdx.z;
  const bf16* xb = xT + (size_t)b * NN * CC;

  v8f acc[4] = {};
  int m = mt * 16 + li;

  for (int kb = 0; kb < CC; kb += 32) {
    // A-tile: row m, K = kb + {g*8..g*8+7} U {g*8+16..g*8+23}
    const float* wrow = Wmat + (size_t)m * CC + kb + g * 8;
    v4f a0 = *(const v4f*)(wrow);
    v4f a1 = *(const v4f*)(wrow + 4);
    v4f a2 = *(const v4f*)(wrow + 16);
    v4f a3 = *(const v4f*)(wrow + 20);
    V16 A;
#pragma unroll
    for (int i = 0; i < 4; ++i) {
      A.v[i]      = (bf16)a0[i];
      A.v[4 + i]  = (bf16)a1[i];
      A.v[8 + i]  = (bf16)a2[i];
      A.v[12 + i] = (bf16)a3[i];
    }
#pragma unroll
    for (int t = 0; t < 4; ++t) {
      int n = nb * 64 + t * 16 + li;
      const bf16* brow = xb + (size_t)n * CC + kb + g * 8;
      V16 Bv;
      Bv.h[0] = *(const v8bf*)(brow);
      Bv.h[1] = *(const v8bf*)(brow + 16);
      acc[t] = __builtin_amdgcn_wmma_f32_16x16x32_bf16(
          false, A.v, false, Bv.v, (short)0, acc[t], false, false);
    }
  }

  bf16* ob = out + (size_t)b * M * NN;
#pragma unroll
  for (int t = 0; t < 4; ++t)
#pragma unroll
    for (int r = 0; r < 8; ++r) {
      int mm = mt * 16 + r + 8 * g;
      int nn = nb * 64 + t * 16 + li;
      float v = (acc[t][r] + bias[mm]) * ascale;
      ob[(size_t)mm * sm + (size_t)nn * sn] = (bf16)v;
    }
}

// ---------------------------------------------------------------------------
// Kernel 3: flash attention. O(N x C) = softmax(Q K) * V^T  per batch.
// Q pre-scaled by log2(e); softmax in base-2 (native v_exp_f32 domain).
// Workgroup = 16 queries, 4 waves x 128 value channels, 64-key blocks.
// ---------------------------------------------------------------------------
__global__ __launch_bounds__(128) void flash_attn(
    const bf16* __restrict__ Q, const bf16* __restrict__ K,
    const bf16* __restrict__ V, float* __restrict__ O) {
  __shared__ bf16 Plds[4][16 * KB];
  int lane = threadIdx.x & 31, wid = threadIdx.x >> 5;
  int g = lane >> 4, li = lane & 15;
  int q0 = blockIdx.x * 16;
  int b  = blockIdx.y;
  int ch0 = wid * 128;
  const bf16* Qb = Q + (size_t)b * NN * DD;
  const bf16* Kb = K + (size_t)b * NN * DD;
  const bf16* Vb = V + (size_t)b * CC * NN;

  // Q A-tiles (16 queries x 64 d), two K=32 chunks, resident in VGPRs
  V16 Qa[2];
  {
    const bf16* qrow = Qb + (size_t)(q0 + li) * DD;
#pragma unroll
    for (int dc = 0; dc < 2; ++dc) {
      const bf16* p = qrow + dc * 32 + g * 8;
      Qa[dc].h[0] = *(const v8bf*)(p);
      Qa[dc].h[1] = *(const v8bf*)(p + 16);
    }
  }

  v8f acc[8] = {};
  float m_run[8], l_run[8];
#pragma unroll
  for (int r = 0; r < 8; ++r) { m_run[r] = -3.0e38f; l_run[r] = 0.0f; }

  for (int kb = 0; kb < NN; kb += KB) {
    // ---- S = Q * K  (16 x 64 keys), four 16-key column tiles ----
    v8f S[4];
#pragma unroll
    for (int st = 0; st < 4; ++st) {
      int key = kb + st * 16 + li;
      const bf16* krow = Kb + (size_t)key * DD + g * 8;
      V16 B0, B1;
      B0.h[0] = *(const v8bf*)(krow);
      B0.h[1] = *(const v8bf*)(krow + 16);
      B1.h[0] = *(const v8bf*)(krow + 32);
      B1.h[1] = *(const v8bf*)(krow + 48);
      v8f s = {};
      s = __builtin_amdgcn_wmma_f32_16x16x32_bf16(false, Qa[0].v, false, B0.v,
                                                  (short)0, s, false, false);
      s = __builtin_amdgcn_wmma_f32_16x16x32_bf16(false, Qa[1].v, false, B1.v,
                                                  (short)0, s, false, false);
      S[st] = s;
    }

    // ---- block row max (row = r + 8*g, cols across 16 lanes) ----
    float mblk[8];
#pragma unroll
    for (int r = 0; r < 8; ++r)
      mblk[r] = fmaxf(fmaxf(S[0][r], S[1][r]), fmaxf(S[2][r], S[3][r]));
#pragma unroll
    for (int off = 1; off < 16; off <<= 1)
#pragma unroll
      for (int r = 0; r < 8; ++r)
        mblk[r] = fmaxf(mblk[r], __shfl_xor(mblk[r], off, 32));

    bool grow = false;
#pragma unroll
    for (int r = 0; r < 8; ++r) grow |= (mblk[r] > m_run[r]);

    float mnew[8];
    if (__any((int)grow)) {
      // new row max somewhere: rescale running state (wave-uniform branch)
      float scale[8];
#pragma unroll
      for (int r = 0; r < 8; ++r) {
        mnew[r]  = fmaxf(m_run[r], mblk[r]);
        scale[r] = fast_exp2(m_run[r] - mnew[r]);
        l_run[r] *= scale[r];
        m_run[r] = mnew[r];
      }
#pragma unroll
      for (int t = 0; t < 8; ++t)
#pragma unroll
        for (int r = 0; r < 8; ++r) acc[t][r] *= scale[r];
    } else {
#pragma unroll
      for (int r = 0; r < 8; ++r) mnew[r] = m_run[r];
    }

    // ---- P = exp2(S - m), row sums ----
    float psum[8];
#pragma unroll
    for (int r = 0; r < 8; ++r) {
      S[0][r] = fast_exp2(S[0][r] - mnew[r]);
      S[1][r] = fast_exp2(S[1][r] - mnew[r]);
      S[2][r] = fast_exp2(S[2][r] - mnew[r]);
      S[3][r] = fast_exp2(S[3][r] - mnew[r]);
      psum[r] = (S[0][r] + S[1][r]) + (S[2][r] + S[3][r]);
    }
#pragma unroll
    for (int off = 1; off < 16; off <<= 1)
#pragma unroll
      for (int r = 0; r < 8; ++r) psum[r] += __shfl_xor(psum[r], off, 32);
#pragma unroll
    for (int r = 0; r < 8; ++r) l_run[r] += psum[r];

    // ---- P: D-layout -> A-layout via per-wave LDS buffer ----
    bf16* pb = &Plds[wid][0];
#pragma unroll
    for (int st = 0; st < 4; ++st)
#pragma unroll
      for (int r = 0; r < 8; ++r)
        pb[(r + 8 * g) * KB + st * 16 + li] = (bf16)S[st][r];
    V16 Pa[2];
#pragma unroll
    for (int pc = 0; pc < 2; ++pc) {
      const bf16* pr = pb + li * KB + pc * 32 + g * 8;
      Pa[pc].h[0] = *(const v8bf*)(pr);
      Pa[pc].h[1] = *(const v8bf*)(pr + 16);
    }

    // ---- O += P(16x64) * V^T(64 x 128 ch), 8 channel tiles x 2 chunks ----
#pragma unroll
    for (int t = 0; t < 8; ++t) {
      int e = ch0 + t * 16 + li;
      const bf16* vrow = Vb + (size_t)e * NN + kb + g * 8;
      V16 Bv0, Bv1;
      Bv0.h[0] = *(const v8bf*)(vrow);
      Bv0.h[1] = *(const v8bf*)(vrow + 16);
      Bv1.h[0] = *(const v8bf*)(vrow + 32);
      Bv1.h[1] = *(const v8bf*)(vrow + 48);
      acc[t] = __builtin_amdgcn_wmma_f32_16x16x32_bf16(
          false, Pa[0].v, false, Bv0.v, (short)0, acc[t], false, false);
      acc[t] = __builtin_amdgcn_wmma_f32_16x16x32_bf16(
          false, Pa[1].v, false, Bv1.v, (short)0, acc[t], false, false);
    }
  }

  // ---- finalize: O[q, e] = acc / l ----
  float* Ob = O + (size_t)b * NN * CC;
  float inv[8];
#pragma unroll
  for (int r = 0; r < 8; ++r) inv[r] = 1.0f / l_run[r];
#pragma unroll
  for (int t = 0; t < 8; ++t)
#pragma unroll
    for (int r = 0; r < 8; ++r) {
      int row = q0 + r + 8 * g;
      int e   = ch0 + t * 16 + li;
      Ob[(size_t)row * CC + e] = acc[t][r] * inv[r];
    }
}

// ---------------------------------------------------------------------------
// Kernel 4: out(B,C,N) = gamma * O^T + x   (LDS tile transpose)
// ---------------------------------------------------------------------------
__global__ void epilogue(const float* __restrict__ Obuf,
                         const float* __restrict__ x,
                         const float* __restrict__ gamma,
                         float* __restrict__ out) {
  __shared__ float tile[16][17];
  int b  = blockIdx.z;
  int c0 = blockIdx.x * 16, n0 = blockIdx.y * 16;
  int tx = threadIdx.x, ty = threadIdx.y;
  const float* Ob = Obuf + (size_t)b * NN * CC;
  tile[ty][tx] = Ob[(size_t)(n0 + ty) * CC + c0 + tx];
  __syncthreads();
  float gm = gamma[0];
  size_t idx = (size_t)b * CC * NN + (size_t)(c0 + ty) * NN + n0 + tx;
  out[idx] = gm * tile[tx][ty] + x[idx];
}

// ---------------------------------------------------------------------------
extern "C" void kernel_launch(void* const* d_in, const int* in_sizes, int n_in,
                              void* d_out, int out_size, void* d_ws, size_t ws_size,
                              hipStream_t stream) {
  const float* x     = (const float*)d_in[0];
  const float* Wq    = (const float*)d_in[1];
  const float* bq    = (const float*)d_in[2];
  const float* Wk    = (const float*)d_in[3];
  const float* bk    = (const float*)d_in[4];
  const float* Wv    = (const float*)d_in[5];
  const float* bv    = (const float*)d_in[6];
  const float* gamma = (const float*)d_in[7];
  float* out = (float*)d_out;

  char* ws = (char*)d_ws;
  size_t off = 0;
  bf16* xT = (bf16*)(ws + off); off += (size_t)BB * NN * CC * 2;  // 16 MB
  bf16* Qb = (bf16*)(ws + off); off += (size_t)BB * NN * DD * 2;  //  2 MB
  bf16* Kb = (bf16*)(ws + off); off += (size_t)BB * NN * DD * 2;  //  2 MB
  bf16* Vb = (bf16*)(ws + off); off += (size_t)BB * CC * NN * 2;  // 16 MB
  float* Ob = (float*)(ws + off);                                  // 32 MB

  const float LOG2E = 1.4426950408889634f;
  dim3 tb(16, 16);
  xpose_bf16<<<dim3(CC / 16, NN / 16, BB), tb, 0, stream>>>(x, xT);
  proj_gemm<<<dim3(DD / 16, NN / 64, BB), 32, 0, stream>>>(Wq, bq, xT, Qb, DD, 1, DD, LOG2E);
  proj_gemm<<<dim3(DD / 16, NN / 64, BB), 32, 0, stream>>>(Wk, bk, xT, Kb, DD, 1, DD, 1.0f);
  proj_gemm<<<dim3(CC / 16, NN / 64, BB), 32, 0, stream>>>(Wv, bv, xT, Vb, CC, NN, 1, 1.0f);
  flash_attn<<<dim3(NN / 16, BB), 128, 0, stream>>>(Qb, Kb, Vb, Ob);
  epilogue<<<dim3(CC / 16, NN / 16, BB), tb, 0, stream>>>(Ob, x, gamma, out);
}
